// CTEG_official_60790967107644
// MI455X (gfx1250) — matrix operand order repaired and checked by reading, not compiled, not run on hardware
//
#include <hip/hip_runtime.h>
#include <hip/hip_bf16.h>
#include <math.h>

// ---------------- problem constants ----------------
#define BB 32
#define TT_TOPIC 5
#define TT_ESSAY 60
#define MEMN 120
#define ED 300
#define HD 512
#define AD 128
#define VD 50000
#define G4H 2048           // 4*HD
#define LID 1624           // ED + HD + ED + HD  (x, attn, mt, h)

typedef __bf16 v16bf __attribute__((ext_vector_type(16)));
typedef float  v8f  __attribute__((ext_vector_type(8)));

__device__ __forceinline__ float sigm(float x) { return 1.f / (1.f + expf(-x)); }

// ======================================================================
// Generic WMMA GEMM:  C[M,N] = A[M,K] @ B[N,K]^T + bias[N]  (act: 0=none 1=tanh)
// A is f32 row-major (converted to bf16 at load).  B is f32 (WBF16=false) or
// pre-converted bf16 (WBF16=true), row-major [N,K].
// Requirements: M % 32 == 0.  N, K arbitrary (N-edge / K-tail handled OUTSIDE
// the hot loop so the steady state is loads + cvt + 4 x wmma only).
// Each wave computes a 32x32 block of C.
// ======================================================================
__device__ __forceinline__ v16bf load_a_fast(const float* __restrict__ arow,
                                             int ka, int kb) {
  v16bf a;
#pragma unroll
  for (int e = 0; e < 8; ++e) {
    a[e]     = (__bf16)arow[ka + e];
    a[8 + e] = (__bf16)arow[kb + e];
  }
  return a;
}

__device__ __forceinline__ v16bf load_a_tail(const float* __restrict__ arow,
                                             int ka, int kb, int K) {
  v16bf a;
#pragma unroll
  for (int e = 0; e < 8; ++e) {
    a[e]     = (__bf16)((ka + e < K) ? arow[ka + e] : 0.f);
    a[8 + e] = (__bf16)((kb + e < K) ? arow[kb + e] : 0.f);
  }
  return a;
}

template <bool WBF16>
__device__ __forceinline__ v16bf load_b_fast(const void* __restrict__ Bv,
                                             int nB, int K, int kb) {
  v16bf b;
  if constexpr (WBF16) {
    const __bf16* brow = (const __bf16*)Bv + (size_t)nB * K;
#pragma unroll
    for (int e = 0; e < 16; ++e) b[e] = brow[kb + e];
  } else {
    const float* brow = (const float*)Bv + (size_t)nB * K;
#pragma unroll
    for (int e = 0; e < 16; ++e) b[e] = (__bf16)brow[kb + e];
  }
  return b;
}

template <bool WBF16>
__device__ __forceinline__ v16bf load_b_guard(const void* __restrict__ Bv,
                                              int nB, int N, int K, int kb) {
  v16bf b;
  const bool ok = nB < N;
  if constexpr (WBF16) {
    const __bf16* brow = (const __bf16*)Bv + (size_t)nB * K;
#pragma unroll
    for (int e = 0; e < 16; ++e)
      b[e] = (ok && (kb + e) < K) ? brow[kb + e] : (__bf16)0.f;
  } else {
    const float* brow = (const float*)Bv + (size_t)nB * K;
#pragma unroll
    for (int e = 0; e < 16; ++e)
      b[e] = (__bf16)((ok && (kb + e) < K) ? brow[kb + e] : 0.f);
  }
  return b;
}

template <bool WBF16>
__global__ __launch_bounds__(128) void k_gemm(const float* __restrict__ A,
                                              const void* __restrict__ Bv,
                                              const float* __restrict__ bias,
                                              float* __restrict__ C,
                                              int M, int N, int K, int act) {
  const int lane = threadIdx.x & 31;
  const int wave = threadIdx.x >> 5;
  const int m0 = blockIdx.y * 32;
  const int n0 = blockIdx.x * 128 + wave * 32;
  if (n0 >= N) return;                      // whole wave out of range
  const int mA = lane & 15;                 // A row in tile / B col in tile / C col
  const int h  = lane >> 4;

  const float* arow0 = A + (size_t)(m0 + mA) * K;
  const float* arow1 = A + (size_t)(m0 + 16 + mA) * K;
  const int nB0 = n0 + mA;
  const int nB1 = n0 + 16 + mA;

  v8f acc00 = {}, acc01 = {}, acc10 = {}, acc11 = {};

  const int Kmain = K & ~31;
  const bool inner = (n0 + 32 <= N);        // all 32 columns of this wave valid

  if (inner) {
    // ---- guard-free hot loop: loads + cvt + 4 wmma ----
#pragma unroll 2
    for (int k0 = 0; k0 < Kmain; k0 += 32) {
      if (k0 + 128 < Kmain) {
        __builtin_prefetch(arow0 + k0 + 128, 0, 1);   // global_prefetch_b8
        __builtin_prefetch(arow1 + k0 + 128, 0, 1);
      }
      const int ka = k0 + h * 16;
      const int kb = k0 + h * 16 + 8;
      v16bf a0 = load_a_fast(arow0, ka, kb);
      v16bf a1 = load_a_fast(arow1, ka, kb);
      const int kB = k0 + h * 16;
      v16bf b0 = load_b_fast<WBF16>(Bv, nB0, K, kB);
      v16bf b1 = load_b_fast<WBF16>(Bv, nB1, K, kB);
      acc00 = __builtin_amdgcn_wmma_f32_16x16x32_bf16(false, a0, false, b0, (short)0, acc00, false, false);
      acc01 = __builtin_amdgcn_wmma_f32_16x16x32_bf16(false, a0, false, b1, (short)0, acc01, false, false);
      acc10 = __builtin_amdgcn_wmma_f32_16x16x32_bf16(false, a1, false, b0, (short)0, acc10, false, false);
      acc11 = __builtin_amdgcn_wmma_f32_16x16x32_bf16(false, a1, false, b1, (short)0, acc11, false, false);
    }
  } else {
    // ---- N-edge waves only ----
    for (int k0 = 0; k0 < Kmain; k0 += 32) {
      const int ka = k0 + h * 16;
      const int kb = k0 + h * 16 + 8;
      v16bf a0 = load_a_fast(arow0, ka, kb);
      v16bf a1 = load_a_fast(arow1, ka, kb);
      const int kB = k0 + h * 16;
      v16bf b0 = load_b_guard<WBF16>(Bv, nB0, N, K, kB);
      v16bf b1 = load_b_guard<WBF16>(Bv, nB1, N, K, kB);
      acc00 = __builtin_amdgcn_wmma_f32_16x16x32_bf16(false, a0, false, b0, (short)0, acc00, false, false);
      acc01 = __builtin_amdgcn_wmma_f32_16x16x32_bf16(false, a0, false, b1, (short)0, acc01, false, false);
      acc10 = __builtin_amdgcn_wmma_f32_16x16x32_bf16(false, a1, false, b0, (short)0, acc10, false, false);
      acc11 = __builtin_amdgcn_wmma_f32_16x16x32_bf16(false, a1, false, b1, (short)0, acc11, false, false);
    }
  }

  if (Kmain < K) {
    // ---- single K-tail iteration, fully guarded ----
    const int k0 = Kmain;
    const int ka = k0 + h * 16;
    const int kb = k0 + h * 16 + 8;
    v16bf a0 = load_a_tail(arow0, ka, kb, K);
    v16bf a1 = load_a_tail(arow1, ka, kb, K);
    const int kB = k0 + h * 16;
    v16bf b0 = load_b_guard<WBF16>(Bv, nB0, N, K, kB);
    v16bf b1 = load_b_guard<WBF16>(Bv, nB1, N, K, kB);
    acc00 = __builtin_amdgcn_wmma_f32_16x16x32_bf16(false, a0, false, b0, (short)0, acc00, false, false);
    acc01 = __builtin_amdgcn_wmma_f32_16x16x32_bf16(false, a0, false, b1, (short)0, acc01, false, false);
    acc10 = __builtin_amdgcn_wmma_f32_16x16x32_bf16(false, a1, false, b0, (short)0, acc10, false, false);
    acc11 = __builtin_amdgcn_wmma_f32_16x16x32_bf16(false, a1, false, b1, (short)0, acc11, false, false);
  }

  const int cn = mA;
  const int rowoff = h * 8;
  auto store_tile = [&](const v8f& acc, int mb, int nb) {
    const int n = nb + cn;
    if (n >= N) return;
    const float bb = bias ? bias[n] : 0.f;
#pragma unroll
    for (int r = 0; r < 8; ++r) {
      float x = acc[r] + bb;
      if (act == 1) x = tanhf(x);
      C[(size_t)(mb + rowoff + r) * N + n] = x;
    }
  };
  store_tile(acc00, m0, n0);
  store_tile(acc01, m0, n0 + 16);
  store_tile(acc10, m0 + 16, n0);
  store_tile(acc11, m0 + 16, n0 + 16);
}

// ---------------------------------------------------------------------
// Elementwise / small kernels
// ---------------------------------------------------------------------
__global__ void k_cvt_bf16(const float* __restrict__ in, __bf16* __restrict__ out, int n) {
  int i = blockIdx.x * blockDim.x + threadIdx.x;
  if (i < n) out[i] = (__bf16)in[i];
}

__global__ void k_vadd(const float* __restrict__ a, const float* __restrict__ b,
                       float* __restrict__ o, int n) {
  int i = blockIdx.x * blockDim.x + threadIdx.x;
  if (i < n) o[i] = a[i] + b[i];
}

__global__ void k_gather(const float* __restrict__ table, const int* __restrict__ idx,
                         float* __restrict__ out, int rows) {
  int i = blockIdx.x * blockDim.x + threadIdx.x;
  if (i >= rows * ED) return;
  int r = i / ED, e = i - r * ED;
  out[i] = table[(size_t)idx[r] * ED + e];
}

// Concatenate decoder weights: Wcat[2048,1624] = [dec_wih | dec_whh]
__global__ void k_concat_w(const float* __restrict__ wih, const float* __restrict__ whh,
                           float* __restrict__ wcat) {
  int i = blockIdx.x * blockDim.x + threadIdx.x;
  if (i >= G4H * LID) return;
  int r = i / LID, cidx = i - r * LID;
  wcat[i] = (cidx < 1112) ? wih[(size_t)r * 1112 + cidx]
                          : whh[(size_t)r * HD + (cidx - 1112)];
}

// ---------------------------------------------------------------------
// Encoder: 5-step LSTM scan for one direction per block (grid.x = 2).
// xW (per-step input contribution incl. both biases) is precomputed via WMMA GEMM.
// h lives in 64KB LDS; c lives in registers of the owning thread.
// ---------------------------------------------------------------------
__global__ __launch_bounds__(1024) void k_encoder(
    const float* __restrict__ xW_fw, const float* __restrict__ xW_bw,
    const float* __restrict__ whh_fw, const float* __restrict__ whh_bw,
    float* __restrict__ hsteps,   // [2][160][512]
    float* __restrict__ hfin,     // [2][32*512]
    float* __restrict__ cfin) {   // [2][32*512]
  const int d = blockIdx.x;
  const float* xW  = d ? xW_bw : xW_fw;
  const float* whh = d ? whh_bw : whh_fw;
  __shared__ float hsh[BB * HD];  // 64KB
  float creg[16], hnew[16];
  const int tid = threadIdx.x;
#pragma unroll
  for (int ii = 0; ii < 16; ++ii) { hsh[tid + 1024 * ii] = 0.f; creg[ii] = 0.f; }
  __syncthreads();

  for (int s = 0; s < TT_TOPIC; ++s) {
    const int tt = d ? (TT_TOPIC - 1 - s) : s;
    for (int ii = 0; ii < 16; ++ii) {
      const int p = tid + 1024 * ii;
      const int b = p >> 9, j = p & 511;
      const float* xg = xW + (size_t)(b * TT_TOPIC + tt) * G4H;
      const float* hrow = hsh + (b << 9);
      float gi = xg[j], gf = xg[HD + j], gg = xg[2 * HD + j], go = xg[3 * HD + j];
      const float* wi  = whh + (size_t)j * HD;
      const float* wf  = whh + (size_t)(HD + j) * HD;
      const float* wgp = whh + (size_t)(2 * HD + j) * HD;
      const float* wo  = whh + (size_t)(3 * HD + j) * HD;
      for (int k = 0; k < HD; ++k) {
        const float hk = hrow[k];
        gi = fmaf(wi[k], hk, gi);
        gf = fmaf(wf[k], hk, gf);
        gg = fmaf(wgp[k], hk, gg);
        go = fmaf(wo[k], hk, go);
      }
      const float c2 = sigm(gf) * creg[ii] + sigm(gi) * tanhf(gg);
      creg[ii] = c2;
      hnew[ii] = sigm(go) * tanhf(c2);
    }
    __syncthreads();  // all reads of old h done
    for (int ii = 0; ii < 16; ++ii) {
      const int p = tid + 1024 * ii;
      hsh[p] = hnew[ii];
      const int b = p >> 9, j = p & 511;
      hsteps[((size_t)d * (BB * TT_TOPIC) + b * TT_TOPIC + tt) * HD + j] = hnew[ii];
    }
    __syncthreads();
  }
#pragma unroll
  for (int ii = 0; ii < 16; ++ii) {
    const int p = tid + 1024 * ii;
    hfin[d * (BB * HD) + p] = hsh[p];
    cfin[d * (BB * HD) + p] = creg[ii];
  }
}

// enc_outs = fw + bw ; h0 = h_fw + h_bw ; c0 = c_fw + c_bw
__global__ void k_combine_enc(const float* __restrict__ hsteps,
                              const float* __restrict__ hfin, const float* __restrict__ cfin,
                              float* __restrict__ enc_outs,
                              float* __restrict__ h, float* __restrict__ c) {
  int i = blockIdx.x * blockDim.x + threadIdx.x;
  const int NE = BB * TT_TOPIC * HD;
  if (i < NE) enc_outs[i] = hsteps[i] + hsteps[NE + i];
  if (i < BB * HD) {
    h[i] = hfin[i] + hfin[BB * HD + i];
    c[i] = cfin[i] + cfin[BB * HD + i];
  }
}

// ---------------------------------------------------------------------
// Memory read:  sim = softmax(mem . v), mt = sim @ mem,  gate = sigmoid(mem . x)
// one block per batch element
// ---------------------------------------------------------------------
__global__ __launch_bounds__(128) void k_memread(const float* __restrict__ mem,
                                                 const float* __restrict__ v,
                                                 const float* __restrict__ xemb, int t,
                                                 float* __restrict__ mt,
                                                 float* __restrict__ gate) {
  const int b = blockIdx.x;
  const int tid = threadIdx.x;
  const float* vb = v + b * ED;
  const float* xb = xemb + (size_t)(b * TT_ESSAY + t) * ED;
  __shared__ float s_sim[MEMN];
  __shared__ float s_gd[MEMN];
  __shared__ float red[128];

  for (int m = tid; m < MEMN; m += 128) {
    const float* mr = mem + ((size_t)b * MEMN + m) * ED;
    float sv = 0.f, sx = 0.f;
    for (int e = 0; e < ED; ++e) {
      const float me = mr[e];
      sv = fmaf(me, vb[e], sv);
      sx = fmaf(me, xb[e], sx);
    }
    s_sim[m] = sv;
    s_gd[m] = sx;
  }
  __syncthreads();
  // softmax over MEMN
  float mx = -1e30f;
  for (int m = tid; m < MEMN; m += 128) mx = fmaxf(mx, s_sim[m]);
  red[tid] = mx; __syncthreads();
  for (int off = 64; off > 0; off >>= 1) {
    if (tid < off) red[tid] = fmaxf(red[tid], red[tid + off]);
    __syncthreads();
  }
  mx = red[0]; __syncthreads();
  float sm = 0.f;
  for (int m = tid; m < MEMN; m += 128) {
    const float e = expf(s_sim[m] - mx);
    s_sim[m] = e;
    sm += e;
  }
  red[tid] = sm; __syncthreads();
  for (int off = 64; off > 0; off >>= 1) {
    if (tid < off) red[tid] += red[tid + off];
    __syncthreads();
  }
  const float inv = 1.f / red[0];
  __syncthreads();
  for (int m = tid; m < MEMN; m += 128) {
    s_sim[m] *= inv;
    gate[b * MEMN + m] = sigm(s_gd[m]);
  }
  __syncthreads();
  for (int e = tid; e < ED; e += 128) {
    float acc = 0.f;
    for (int m = 0; m < MEMN; ++m)
      acc = fmaf(s_sim[m], mem[((size_t)b * MEMN + m) * ED + e], acc);
    mt[b * ED + e] = acc;
  }
}

// ---------------------------------------------------------------------
// Bahdanau attention over the 5 encoder outputs. One block (128 thr) per b.
// ---------------------------------------------------------------------
__global__ __launch_bounds__(128) void k_attention(const float* __restrict__ q,
                                                   const float* __restrict__ enc_proj,
                                                   const float* __restrict__ enc_outs,
                                                   const float* __restrict__ att_v,
                                                   float* __restrict__ attn) {
  const int b = blockIdx.x;
  const int tid = threadIdx.x;  // == attention dim index (AD == 128)
  __shared__ float sc[TT_TOPIC];
  __shared__ float red[TT_TOPIC][128];
  const float qa = q[b * AD + tid];
  const float av = att_v[tid];
#pragma unroll
  for (int tt = 0; tt < TT_TOPIC; ++tt)
    red[tt][tid] = av * tanhf(enc_proj[(size_t)(b * TT_TOPIC + tt) * AD + tid] + qa);
  __syncthreads();
  for (int off = 64; off > 0; off >>= 1) {
    if (tid < off)
#pragma unroll
      for (int tt = 0; tt < TT_TOPIC; ++tt) red[tt][tid] += red[tt][tid + off];
    __syncthreads();
  }
  if (tid == 0) {
    float mxs = red[0][0];
#pragma unroll
    for (int tt = 1; tt < TT_TOPIC; ++tt) mxs = fmaxf(mxs, red[tt][0]);
    float sm = 0.f;
#pragma unroll
    for (int tt = 0; tt < TT_TOPIC; ++tt) { sc[tt] = expf(red[tt][0] - mxs); sm += sc[tt]; }
    const float inv = 1.f / sm;
#pragma unroll
    for (int tt = 0; tt < TT_TOPIC; ++tt) sc[tt] *= inv;
  }
  __syncthreads();
  for (int hh = tid; hh < HD; hh += 128) {
    float acc = 0.f;
#pragma unroll
    for (int tt = 0; tt < TT_TOPIC; ++tt)
      acc = fmaf(sc[tt], enc_outs[(size_t)(b * TT_TOPIC + tt) * HD + hh], acc);
    attn[b * HD + hh] = acc;
  }
}

// li2[b] = [x_t(300); attn(512); mt(300); h(512)]
__global__ void k_build_li(const float* __restrict__ xemb, int t,
                           const float* __restrict__ attn, const float* __restrict__ mt,
                           const float* __restrict__ h, float* __restrict__ li) {
  int i = blockIdx.x * blockDim.x + threadIdx.x;
  if (i >= BB * LID) return;
  int b = i / LID, j = i - b * LID;
  float val;
  if (j < 300)       val = xemb[(size_t)(b * TT_ESSAY + t) * ED + j];
  else if (j < 812)  val = attn[b * HD + (j - 300)];
  else if (j < 1112) val = mt[b * ED + (j - 812)];
  else               val = h[b * HD + (j - 1112)];
  li[i] = val;
}

// decoder LSTM cell: consume gates, update h/c, record h into Hall[b*60+t]
__global__ void k_lstm_cell(const float* __restrict__ g, float* __restrict__ h,
                            float* __restrict__ c, float* __restrict__ Hall, int t) {
  int i = blockIdx.x * blockDim.x + threadIdx.x;
  if (i >= BB * HD) return;
  int b = i >> 9, j = i & 511;
  const float* gb = g + (size_t)b * G4H;
  const float gi = gb[j], gf = gb[HD + j], gg = gb[2 * HD + j], go = gb[3 * HD + j];
  const float c2 = sigm(gf) * c[i] + sigm(gi) * tanhf(gg);
  const float h2 = sigm(go) * tanhf(c2);
  c[i] = c2;
  h[i] = h2;
  Hall[(size_t)(b * TT_ESSAY + t) * HD + j] = h2;
}

// gated memory write: mem = (1-gate)*mem + gate*(xip1_t + cand_lin), inf->eps
__global__ void k_memupdate(float* __restrict__ mem, const float* __restrict__ gate,
                            const float* __restrict__ cand_lin,
                            const float* __restrict__ xip1, int t) {
  int i = blockIdx.x * blockDim.x + threadIdx.x;
  if (i >= BB * MEMN * ED) return;
  int b = i / (MEMN * ED);
  int rm = i - b * (MEMN * ED);
  int m = rm / ED, e = rm - m * ED;
  const float g = gate[b * MEMN + m];
  const float cand = cand_lin[i] + xip1[(size_t)(b * TT_ESSAY + t) * ED + e];
  float val = (1.f - g) * mem[i] + g * cand;
  if (isinf(val)) val = 1e-40f;
  mem[i] = val;
}

// ---------------------------------------------------------------------
// Host-side orchestration
// ---------------------------------------------------------------------
static inline void gemm_f32w(const float* A, const float* B, const float* bias, float* C,
                             int M, int N, int K, int act, hipStream_t s) {
  dim3 g((N + 127) / 128, M / 32);
  k_gemm<false><<<g, dim3(128), 0, s>>>(A, (const void*)B, bias, C, M, N, K, act);
}
static inline void gemm_bf16w(const float* A, const __bf16* B, const float* bias, float* C,
                              int M, int N, int K, int act, hipStream_t s) {
  dim3 g((N + 127) / 128, M / 32);
  k_gemm<true><<<g, dim3(128), 0, s>>>(A, (const void*)B, bias, C, M, N, K, act);
}
static inline int cdiv(int a, int b) { return (a + b - 1) / b; }

extern "C" void kernel_launch(void* const* d_in, const int* in_sizes, int n_in,
                              void* d_out, int out_size, void* d_ws, size_t ws_size,
                              hipStream_t stream) {
  (void)in_sizes; (void)n_in; (void)out_size; (void)ws_size;
  const float* embed_w = (const float*)d_in[0];
  const float* fw_wih  = (const float*)d_in[1];
  const float* fw_whh  = (const float*)d_in[2];
  const float* fw_bih  = (const float*)d_in[3];
  const float* fw_bhh  = (const float*)d_in[4];
  const float* bw_wih  = (const float*)d_in[5];
  const float* bw_whh  = (const float*)d_in[6];
  const float* bw_bih  = (const float*)d_in[7];
  const float* bw_bhh  = (const float*)d_in[8];
  const float* hp1_w   = (const float*)d_in[9];
  const float* hp1_b   = (const float*)d_in[10];
  const float* hp2_w   = (const float*)d_in[11];
  const float* hp2_b   = (const float*)d_in[12];
  const float* ip1_w   = (const float*)d_in[13];
  const float* ip1_b   = (const float*)d_in[14];
  const float* mp_w    = (const float*)d_in[15];
  const float* mp_b    = (const float*)d_in[16];
  const float* eop_w   = (const float*)d_in[17];
  const float* eop_b   = (const float*)d_in[18];
  const float* att_v   = (const float*)d_in[19];
  const float* dec_wih = (const float*)d_in[20];
  const float* dec_whh = (const float*)d_in[21];
  const float* dec_bih = (const float*)d_in[22];
  const float* dec_bhh = (const float*)d_in[23];
  const float* out_w   = (const float*)d_in[24];
  const float* out_b   = (const float*)d_in[25];
  const int*   topic   = (const int*)d_in[26];
  const int*   essay   = (const int*)d_in[28];
  const int*   mems    = (const int*)d_in[30];
  float* logits = (float*)d_out;  // [B, T_ESSAY, V]

  // ---- workspace carve-out ----
  char* base = (char*)d_ws;
  size_t off = 0;
  auto allocf = [&](size_t n) -> float* {
    float* p = (float*)(base + off); off += ((n * 4 + 255) / 256) * 256; return p;
  };
  auto allocb = [&](size_t n) -> __bf16* {
    __bf16* p = (__bf16*)(base + off); off += ((n * 2 + 255) / 256) * 256; return p;
  };
  float* topic_embed = allocf(BB * TT_TOPIC * ED);
  float* essay_embed = allocf(BB * TT_ESSAY * ED);
  float* mem         = allocf(BB * MEMN * ED);
  float* xW_fw       = allocf(BB * TT_TOPIC * G4H);
  float* xW_bw       = allocf(BB * TT_TOPIC * G4H);
  float* bias_fw     = allocf(G4H);
  float* bias_bw     = allocf(G4H);
  float* bias_dec    = allocf(G4H);
  float* hsteps      = allocf(2 * BB * TT_TOPIC * HD);
  float* hfin        = allocf(2 * BB * HD);
  float* cfin        = allocf(2 * BB * HD);
  float* enc_outs    = allocf(BB * TT_TOPIC * HD);
  float* enc_proj    = allocf(BB * TT_TOPIC * AD);
  float* hbuf        = allocf(BB * HD);
  float* cbuf        = allocf(BB * HD);
  float* Hall        = allocf(BB * TT_ESSAY * HD);
  float* xip1        = allocf(BB * TT_ESSAY * ED);
  float* Wcat        = allocf((size_t)G4H * LID);
  float* cand_lin    = allocf(BB * MEMN * ED);
  float* li2         = allocf(BB * LID);
  float* g_dec       = allocf(BB * G4H);
  float* vbuf        = allocf(BB * ED);
  float* qbuf        = allocf(BB * AD);
  float* mtbuf       = allocf(BB * ED);
  float* attnbuf     = allocf(BB * HD);
  float* gatebuf     = allocf(BB * MEMN);
  __bf16* wcat_bf = allocb((size_t)G4H * LID);
  __bf16* outw_bf = allocb((size_t)VD * HD);
  __bf16* mp_bf   = allocb(ED * ED);
  __bf16* hp1_bf  = allocb(ED * HD);
  __bf16* hp2_bf  = allocb(AD * HD);

  const dim3 blk(256);
  // ---- embeddings ----
  k_gather<<<cdiv(BB * TT_TOPIC * ED, 256), blk, 0, stream>>>(embed_w, topic, topic_embed, BB * TT_TOPIC);
  k_gather<<<cdiv(BB * TT_ESSAY * ED, 256), blk, 0, stream>>>(embed_w, essay, essay_embed, BB * TT_ESSAY);
  k_gather<<<cdiv(BB * MEMN * ED, 256), blk, 0, stream>>>(embed_w, mems, mem, BB * MEMN);
  // ---- fused biases & concatenated decoder weight ----
  k_vadd<<<cdiv(G4H, 256), blk, 0, stream>>>(fw_bih, fw_bhh, bias_fw, G4H);
  k_vadd<<<cdiv(G4H, 256), blk, 0, stream>>>(bw_bih, bw_bhh, bias_bw, G4H);
  k_vadd<<<cdiv(G4H, 256), blk, 0, stream>>>(dec_bih, dec_bhh, bias_dec, G4H);
  k_concat_w<<<cdiv(G4H * LID, 256), blk, 0, stream>>>(dec_wih, dec_whh, Wcat);
  // ---- one-time bf16 weight conversions (stay resident in L2) ----
  k_cvt_bf16<<<cdiv(G4H * LID, 256), blk, 0, stream>>>(Wcat, wcat_bf, G4H * LID);
  k_cvt_bf16<<<cdiv(VD * HD, 256), blk, 0, stream>>>(out_w, outw_bf, VD * HD);
  k_cvt_bf16<<<cdiv(ED * ED, 256), blk, 0, stream>>>(mp_w, mp_bf, ED * ED);
  k_cvt_bf16<<<cdiv(ED * HD, 256), blk, 0, stream>>>(hp1_w, hp1_bf, ED * HD);
  k_cvt_bf16<<<cdiv(AD * HD, 256), blk, 0, stream>>>(hp2_w, hp2_bf, AD * HD);
  // ---- encoder: x-contributions via WMMA, then 5-step scan ----
  gemm_f32w(topic_embed, fw_wih, bias_fw, xW_fw, BB * TT_TOPIC, G4H, ED, 0, stream);
  gemm_f32w(topic_embed, bw_wih, bias_bw, xW_bw, BB * TT_TOPIC, G4H, ED, 0, stream);
  k_encoder<<<2, 1024, 0, stream>>>(xW_fw, xW_bw, fw_whh, bw_whh, hsteps, hfin, cfin);
  k_combine_enc<<<cdiv(BB * TT_TOPIC * HD, 256), blk, 0, stream>>>(hsteps, hfin, cfin,
                                                                   enc_outs, hbuf, cbuf);
  gemm_f32w(enc_outs, eop_w, eop_b, enc_proj, BB * TT_TOPIC, AD, HD, 0, stream);
  // loop-invariant x @ ip1^T for all 60 steps
  gemm_f32w(essay_embed, ip1_w, ip1_b, xip1, BB * TT_ESSAY, ED, ED, 0, stream);

  // ---- decoder: 60 sequential steps ----
  for (int t = 0; t < TT_ESSAY; ++t) {
    gemm_bf16w(hbuf, hp1_bf, hp1_b, vbuf, BB, ED, HD, /*tanh*/ 1, stream);
    gemm_bf16w(cbuf, hp2_bf, hp2_b, qbuf, BB, AD, HD, 0, stream);
    k_memread<<<BB, 128, 0, stream>>>(mem, vbuf, essay_embed, t, mtbuf, gatebuf);
    k_attention<<<BB, 128, 0, stream>>>(qbuf, enc_proj, enc_outs, att_v, attnbuf);
    k_build_li<<<cdiv(BB * LID, 256), blk, 0, stream>>>(essay_embed, t, attnbuf, mtbuf, hbuf, li2);
    // memory candidate GEMM on the *old* memory (before update)
    gemm_bf16w(mem, mp_bf, mp_b, cand_lin, BB * MEMN, ED, ED, 0, stream);
    // decoder gates: one fused GEMM over [x; attn; mt; h]
    gemm_bf16w(li2, wcat_bf, bias_dec, g_dec, BB, G4H, LID, 0, stream);
    k_lstm_cell<<<cdiv(BB * HD, 256), blk, 0, stream>>>(g_dec, hbuf, cbuf, Hall, t);
    k_memupdate<<<cdiv(BB * MEMN * ED, 256), blk, 0, stream>>>(mem, gatebuf, cand_lin, xip1, t);
  }

  // ---- batched output projection: all 60 steps in one big WMMA GEMM ----
  // Hall rows are (b*60+t) so C is exactly the [B, T_ESSAY, V] output layout.
  gemm_bf16w(Hall, outw_bf, out_b, logits, BB * TT_ESSAY, VD, HD, 0, stream);
}
